// MCGATE_19713899889141
// MI455X (gfx1250) — compile-verified, matrix operand order checked by reading, not run on hardware
//
#include <hip/hip_runtime.h>
#include <hip/hip_bf16.h>

// ---------------------------------------------------------------------------
// MCGATE graph-attention encoder/decoder for gfx1250 (CDNA5, wave32, WMMA).
// All GEMMs go through v_wmma_f32_16x16x32_f16 (f16 in, f32 accumulate).
// att[H,N,N] kept in f16 (75.5 MB -> resident in the 192 MB L2).
// k_attmm uses the Tensor Data Mover to double-buffer att slabs into LDS.
// ---------------------------------------------------------------------------

#define DEVINL __device__ __forceinline__

typedef __attribute__((ext_vector_type(16))) _Float16 v16h;
typedef __attribute__((ext_vector_type(8)))  _Float16 v8h;
typedef __attribute__((ext_vector_type(8)))  float    v8f;
typedef __attribute__((ext_vector_type(4)))  unsigned int u32x4;
typedef __attribute__((ext_vector_type(8)))  int          i32x8;
typedef __attribute__((ext_vector_type(4)))  int          i32x4;

#ifndef __has_builtin
#define __has_builtin(x) 0
#endif

#if __has_builtin(__builtin_amdgcn_tensor_load_to_lds) && \
    __has_builtin(__builtin_amdgcn_s_wait_tensorcnt)
#define MCG_HAS_TDM 1
#else
#define MCG_HAS_TDM 0
#endif

#if defined(__has_include)
#if __has_include(<hip/amd_detail/amd_gfx1250_TDM.h>)
#define MCG_TDM_6ARG 1   // therock / clang-23 headers -> 6-arg builtin
#else
#define MCG_TDM_6ARG 0   // ROCm 7.2 / clang-22 -> 5-arg builtin
#endif
#else
#define MCG_TDM_6ARG 0
#endif

constexpr int   N_ = 3072;
constexpr int   F_ = 512;
constexpr int   H_ = 4;
constexpr int   D_ = 128;
constexpr int   KSLAB_ = 32;
constexpr int   NSLAB_ = N_ / KSLAB_;   // 96
constexpr float EPS_ = 1e-6f;

DEVINL v8f zero8f() {
  v8f z;
#pragma unroll
  for (int i = 0; i < 8; ++i) z[i] = 0.0f;
  return z;
}

DEVINL v8f wmma32(const v16h& a, const v16h& b, const v8f& c) {
  // D = A(16x32 f16) * B(32x16 f16) + C(16x16 f32)
  return __builtin_amdgcn_wmma_f32_16x16x32_f16(
      /*neg_a=*/false, a, /*neg_b=*/false, b,
      /*c_mod=*/(short)0, c, /*reuse_a=*/false, /*reuse_b=*/false);
}

// A fragment (16x32 f16) from row-major f16 memory, leading dim lda (elems).
// ISA layout: lane = row (mod 16); lanes 0-15 hold K 0-7 & 16-23,
// lanes 16-31 hold K 8-15 & 24-31.
DEVINL v16h load_a_f16(const _Float16* __restrict__ A, int lda) {
  const int lane = threadIdx.x & 31;
  const int row  = lane & 15;
  const int kb   = (lane >> 4) << 3;     // 0 or 8
  const _Float16* p = A + (size_t)row * lda + kb;
  v8h lo = *(const v8h*)(p);
  v8h hi = *(const v8h*)(p + 16);
  v16h r;
#pragma unroll
  for (int i = 0; i < 8; ++i) { r[i] = lo[i]; r[i + 8] = hi[i]; }
  return r;
}

// Same A fragment, but from f32 memory with on-the-fly f16 conversion.
DEVINL v16h load_a_f32(const float* __restrict__ A, int lda) {
  const int lane = threadIdx.x & 31;
  const int row  = lane & 15;
  const int kb   = (lane >> 4) << 3;
  const float* p = A + (size_t)row * lda + kb;
  v16h r;
#pragma unroll
  for (int i = 0; i < 8; ++i) {
    r[i]     = (_Float16)p[i];
    r[i + 8] = (_Float16)p[i + 16];
  }
  return r;
}

// 16x32 fragment from LDS where the "parallel" index (B column / A row) has
// stride `stride` halves and K is contiguous: two 16B ds loads per lane.
DEVINL v16h load_frag_lds(const _Float16* __restrict__ bt, int stride) {
  const int lane = threadIdx.x & 31;
  const int par  = lane & 15;
  const int kb   = (lane >> 4) << 3;
  const _Float16* p = bt + (size_t)par * stride + kb;
  v8h lo = *(const v8h*)(p);
  v8h hi = *(const v8h*)(p + 16);
  v16h r;
#pragma unroll
  for (int i = 0; i < 8; ++i) { r[i] = lo[i]; r[i + 8] = hi[i]; }
  return r;
}

DEVINL float elu1(float x) { return x > 0.0f ? x : __expf(x) - 1.0f; }

#if MCG_HAS_TDM
// Issue a TDM 2D tile load: 128 rows x 32 f16 from a row-major tensor with
// row stride N_ halves, into LDS with hardware padding: 64B row + 16B pad
// -> 80B (40-half) LDS row stride (bank-conflict-free fragment reads).
DEVINL void tdm_load_att_slab(const _Float16* g, unsigned ldsOff) {
  const unsigned long long ga = (unsigned long long)(size_t)g;
  u32x4 g0;
  g0[0] = 1u;                                   // count=1, user mode
  g0[1] = ldsOff;                               // LDS byte address
  g0[2] = (unsigned)ga;                         // global_addr[31:0]
  g0[3] = (unsigned)((ga >> 32) & 0x01FFFFFFu)  // global_addr[56:32]
          | (2u << 30);                         // type=2 ("image")
  i32x8 g1;
  g1[0] = (int)((1u << 16)                      // data_size = 2 bytes
                | (1u << 20)                    // pad_enable
                | (3u << 22)                    // pad_interval: 16 DWORDs
                | (3u << 25));                  // pad_amount: 4 DWORDs
  g1[1] = (int)(((unsigned)N_ & 0xFFFFu) << 16);          // tensor_dim0 lo
  g1[2] = (int)(((unsigned)N_ >> 16) |
                (((unsigned)N_ & 0xFFFFu) << 16));        // dim0 hi | dim1 lo
  g1[3] = (int)(((unsigned)N_ >> 16) |
                ((unsigned)KSLAB_ << 16));                // dim1 hi | tile_dim0
  g1[4] = (int)(128u);                                    // tile_dim1 = 128 rows
  g1[5] = (int)(unsigned)N_;                              // tensor_dim0_stride
  g1[6] = 0;
  g1[7] = 0;
  const i32x4 z4 = {0, 0, 0, 0};
#if MCG_TDM_6ARG
  const i32x8 z8 = {0, 0, 0, 0, 0, 0, 0, 0};
  __builtin_amdgcn_tensor_load_to_lds(g0, g1, z4, z4, z8, 0);
#else
  __builtin_amdgcn_tensor_load_to_lds(g0, g1, z4, z4, 0);
#endif
}
#endif  // MCG_HAS_TDM

// ---------------------------------------------------------------------------
// Per-head GEMM with [D x D] weight:
//   stageTrans=1:  C[n,d] = sum_e A[n, hD+e] * W[h][e][d]      (X @ W)
//   stageTrans=0:  C[n,e] = sum_d A[n, hD+d] * W[h][e][d]      (Y @ W^T)
// A, C32, C16 are [N, F] with head column offset h*D.
// grid = (N/128, H), block = 256 (8 waves, 16 rows each, full 128 cols).
// ---------------------------------------------------------------------------
__global__ __launch_bounds__(256) void k_headgemm(
    const float* __restrict__ A, const float* __restrict__ Wsrc,
    float* __restrict__ C32, _Float16* __restrict__ C16,
    int stageTrans, int doElu)
{
  __shared__ __align__(16) _Float16 bt[D_ * 136];   // BT[col][k], pad 136
  const int h   = blockIdx.y;
  const int tid = threadIdx.x;
  const float* W = Wsrc + (size_t)h * D_ * D_;

  for (int idx = tid; idx < D_ * D_; idx += 256) {
    const int e = idx >> 7, d = idx & 127;
    const _Float16 w = (_Float16)W[idx];
    if (stageTrans) bt[d * 136 + e] = w;   // col=d, k=e
    else            bt[e * 136 + d] = w;   // col=e, k=d
  }
  __syncthreads();

  const int wave = tid >> 5;
  const int m0   = blockIdx.x * 128 + wave * 16;
  const float* Ap = A + (size_t)m0 * F_ + h * D_;

  v8f acc[8];
#pragma unroll
  for (int i = 0; i < 8; ++i) acc[i] = zero8f();

#pragma unroll
  for (int k0 = 0; k0 < D_; k0 += 32) {
    const v16h a = load_a_f32(Ap + k0, F_);
#pragma unroll
    for (int nb = 0; nb < 8; ++nb) {
      const v16h b = load_frag_lds(bt + (size_t)(nb * 16) * 136 + k0, 136);
      acc[nb] = wmma32(a, b, acc[nb]);
    }
  }

  const int lane = tid & 31;
  const int col  = lane & 15;
  const int rb   = (lane >> 4) << 3;
#pragma unroll
  for (int nb = 0; nb < 8; ++nb) {
    const size_t base = (size_t)(m0 + rb) * F_ + h * D_ + nb * 16 + col;
#pragma unroll
    for (int r = 0; r < 8; ++r) {
      float v = acc[nb][r];
      if (doElu) v = elu1(v);
      C32[base + (size_t)r * F_] = v;
      if (C16) C16[base + (size_t)r * F_] = (_Float16)v;
    }
  }
}

// ---------------------------------------------------------------------------
// f1[h,n] = <Xw[h,n,:], V[h,0,:,0]>, f2[h,n] = <Xw[h,n,:], V[h,1,:,0]>
// ---------------------------------------------------------------------------
__global__ __launch_bounds__(256) void k_fvec(
    const float* __restrict__ Xw, const float* __restrict__ V,
    float* __restrict__ f1, float* __restrict__ f2)
{
  const int idx = blockIdx.x * 256 + threadIdx.x;   // h*N + n
  if (idx >= H_ * N_) return;
  const int h = idx / N_, n = idx % N_;
  const float* x  = Xw + (size_t)n * F_ + h * D_;
  const float* v1 = V + (size_t)(h * 2) * D_;
  const float* v2 = v1 + D_;
  float s1 = 0.0f, s2 = 0.0f;
#pragma unroll 4
  for (int d = 0; d < D_; ++d) { s1 += x[d] * v1[d]; s2 += x[d] * v2[d]; }
  f1[idx] = s1; f2[idx] = s2;
}

// ---------------------------------------------------------------------------
// Masked dual softmax + blend -> att f16.  One block per node row i.
// sigmoid outputs are bounded in (0,1) so exp needs no max-subtraction.
// att = 0.5*softmax_long + 0.5*softmax_local (0.5 folded into reciprocals).
// ---------------------------------------------------------------------------
__global__ __launch_bounds__(256) void k_scores(
    const float* __restrict__ f1, const float* __restrict__ f2,
    const float* __restrict__ localp, const float* __restrict__ longp,
    _Float16* __restrict__ att)
{
  __shared__ unsigned char flg[N_];
  __shared__ float sg[N_];
  __shared__ float redA[256], redB[256];
  const int i   = blockIdx.x;
  const int tid = threadIdx.x;

  for (int j = tid; j < N_; j += 256) {
    unsigned char f = 0;
    if (localp[(size_t)i * N_ + j] != 0.0f) f |= 1;
    if (longp [(size_t)i * N_ + j] != 0.0f) f |= 2;
    flg[j] = f;
  }
  __syncthreads();

  for (int h = 0; h < H_; ++h) {
    const float  f1i = f1[h * N_ + i];
    const float* f2h = f2 + h * N_;
    float sl = 0.0f, sgl = 0.0f;
    for (int j = tid; j < N_; j += 256) {
      const float sig = 1.0f / (1.0f + __expf(-(f1i + f2h[j])));
      const float e   = __expf(sig);
      sg[j] = e;
      const unsigned char f = flg[j];
      if (f & 1) sl  += e;
      if (f & 2) sgl += e;
    }
    redA[tid] = sl; redB[tid] = sgl;
    __syncthreads();
    for (int off = 128; off > 0; off >>= 1) {
      if (tid < off) { redA[tid] += redA[tid + off]; redB[tid] += redB[tid + off]; }
      __syncthreads();
    }
    const float invl = redA[0] > 0.0f ? 0.5f / redA[0] : 0.0f;
    const float invg = redB[0] > 0.0f ? 0.5f / redB[0] : 0.0f;
    __syncthreads();

    _Float16* arow = att + (size_t)h * N_ * N_ + (size_t)i * N_;
    for (int j = tid; j < N_; j += 256) {
      const unsigned char f = flg[j];
      const float e = sg[j];
      float a = 0.0f;
      if (f & 1) a += e * invl;
      if (f & 2) a += e * invg;
      arow[j] = (_Float16)a;
    }
    __syncthreads();
  }
}

// ---------------------------------------------------------------------------
// C[n, hD+d] = sum_j att[h,n,j] * B[j, hD+d]   (the 2x 9.7 GFLOP matmuls)
// grid = (N/128, H), block 256.  B staged transposed in LDS per 32-wide
// K slab; att slabs (128x32) streamed into LDS by the Tensor Data Mover
// with double buffering (TENSORcnt split-wait), falling back to direct
// global A-fragment loads when the TDM builtin is unavailable.
// ---------------------------------------------------------------------------
__global__ __launch_bounds__(256) void k_attmm(
    const _Float16* __restrict__ att, const _Float16* __restrict__ B,
    float* __restrict__ C)
{
  __shared__ __align__(16) _Float16 bt[D_ * 40];        // BT[col][k], pad 40
#if MCG_HAS_TDM
  __shared__ __align__(16) _Float16 at[2][128 * 40];    // att slabs, 80B rows
#endif
  const int h    = blockIdx.y;
  const int tid  = threadIdx.x;
  const int wave = tid >> 5;
  const int m0   = blockIdx.x * 128 + wave * 16;

  const _Float16* ablk = att + (size_t)h * N_ * N_ + (size_t)blockIdx.x * 128 * N_;
#if !MCG_HAS_TDM
  const _Float16* ah = att + (size_t)h * N_ * N_ + (size_t)m0 * N_;
#endif

  const int sj   = tid >> 3;          // 0..31 : k row within slab
  const int sseg = (tid & 7) * 16;    // column segment base

  v8f acc[8];
#pragma unroll
  for (int i = 0; i < 8; ++i) acc[i] = zero8f();

#if MCG_HAS_TDM
  const unsigned atOff0 = (unsigned)(size_t)(&at[0][0]);
  const unsigned atOff1 = (unsigned)(size_t)(&at[1][0]);
  if (tid < 32) tdm_load_att_slab(ablk, atOff0);        // prime buffer 0
#endif

  for (int i = 0; i < NSLAB_; ++i) {
    const int k0 = i * KSLAB_;
    {   // stage B slab (transposed) -- all 256 threads
      const _Float16* bp = B + (size_t)(k0 + sj) * F_ + h * D_ + sseg;
      const v8h x0 = *(const v8h*)(bp);
      const v8h x1 = *(const v8h*)(bp + 8);
#pragma unroll
      for (int q = 0; q < 8; ++q) {
        bt[(sseg + q) * 40 + sj]     = x0[q];
        bt[(sseg + 8 + q) * 40 + sj] = x1[q];
      }
    }
#if MCG_HAS_TDM
    if (tid < 32) {
      if (i + 1 < NSLAB_) {
        // stream next slab into the other buffer, then wait for current
        tdm_load_att_slab(ablk + (k0 + KSLAB_), ((i + 1) & 1) ? atOff1 : atOff0);
        __builtin_amdgcn_s_wait_tensorcnt(1);
      } else {
        __builtin_amdgcn_s_wait_tensorcnt(0);
      }
    }
#endif
    __syncthreads();
#if MCG_HAS_TDM
    const v16h a = load_frag_lds(&at[i & 1][wave * 16 * 40], 40);
#else
    const v16h a = load_a_f16(ah + k0, N_);
#endif
#pragma unroll
    for (int nb = 0; nb < 8; ++nb) {
      const v16h b = load_frag_lds(bt + (size_t)(nb * 16) * 40, 40);
      acc[nb] = wmma32(a, b, acc[nb]);
    }
    __syncthreads();
  }

  const int lane = tid & 31;
  const int col  = lane & 15;
  const int rb   = (lane >> 4) << 3;
#pragma unroll
  for (int nb = 0; nb < 8; ++nb) {
    const size_t base = (size_t)(m0 + rb) * F_ + h * D_ + nb * 16 + col;
#pragma unroll
    for (int r = 0; r < 8; ++r) C[base + (size_t)r * F_] = acc[nb][r];
  }
}

// ---------------------------------------------------------------------------
// C[n,o] = sum_i A16[n,i] * Wm[o][i]   (MLP linear, A f16, W f32->f16)
// grid = (N/128, F/128).
// ---------------------------------------------------------------------------
__global__ __launch_bounds__(256) void k_ffgemm(
    const _Float16* __restrict__ A, const float* __restrict__ Wm,
    float* __restrict__ C)
{
  __shared__ __align__(16) _Float16 bt[128 * 40];
  const int tid  = threadIdx.x;
  const int wave = tid >> 5;
  const int m0   = blockIdx.x * 128 + wave * 16;
  const int cb   = blockIdx.y * 128;
  const int scol = tid >> 1;            // 0..127
  const int sseg = (tid & 1) * 16;      // 0 or 16

  v8f acc[8];
#pragma unroll
  for (int i = 0; i < 8; ++i) acc[i] = zero8f();

  for (int k0 = 0; k0 < F_; k0 += 32) {
    {
      const float* wp = Wm + (size_t)(cb + scol) * F_ + k0 + sseg;
      _Float16* lp = bt + scol * 40 + sseg;
#pragma unroll
      for (int i = 0; i < 16; ++i) lp[i] = (_Float16)wp[i];
    }
    __syncthreads();
    const v16h a = load_a_f16(A + (size_t)m0 * F_ + k0, F_);
#pragma unroll
    for (int nb = 0; nb < 8; ++nb) {
      const v16h b = load_frag_lds(bt + (size_t)(nb * 16) * 40, 40);
      acc[nb] = wmma32(a, b, acc[nb]);
    }
    __syncthreads();
  }

  const int lane = tid & 31;
  const int col  = lane & 15;
  const int rb   = (lane >> 4) << 3;
#pragma unroll
  for (int nb = 0; nb < 8; ++nb) {
    const size_t base = (size_t)(m0 + rb) * F_ + cb + nb * 16 + col;
#pragma unroll
    for (int r = 0; r < 8; ++r) C[base + (size_t)r * F_] = acc[nb][r];
  }
}

// ---------------------------------------------------------------------------
// BatchNorm (training stats, biased var) + elementwise kernels.
// ---------------------------------------------------------------------------
__global__ __launch_bounds__(256) void k_bn_stats(
    const float* __restrict__ X, float* __restrict__ mean, float* __restrict__ rstd)
{
  __shared__ float rs[256], rq[256];
  const int col = blockIdx.x, tid = threadIdx.x;
  float s = 0.0f, q = 0.0f;
  for (int n = tid; n < N_; n += 256) {
    const float v = X[(size_t)n * F_ + col];
    s += v; q += v * v;
  }
  rs[tid] = s; rq[tid] = q;
  __syncthreads();
  for (int off = 128; off > 0; off >>= 1) {
    if (tid < off) { rs[tid] += rs[tid + off]; rq[tid] += rq[tid + off]; }
    __syncthreads();
  }
  if (tid == 0) {
    const float m = rs[0] / (float)N_;
    const float var = rq[0] / (float)N_ - m * m;
    mean[col] = m;
    rstd[col] = rsqrtf(var + EPS_);
  }
}

__global__ __launch_bounds__(256) void k_bn_apply(
    const float* __restrict__ X, const float* __restrict__ mean,
    const float* __restrict__ rstd, const float* __restrict__ g,
    const float* __restrict__ b, float* __restrict__ Y, _Float16* __restrict__ Y16)
{
  const int idx = blockIdx.x * 256 + threadIdx.x;
  if (idx >= N_ * F_) return;
  const int col = idx & (F_ - 1);
  const float y = (X[idx] - mean[col]) * rstd[col] * g[col] + b[col];
  Y[idx] = y;
  if (Y16) Y16[idx] = (_Float16)y;
}

__global__ __launch_bounds__(256) void k_elu_res(
    const float* __restrict__ T, const float* __restrict__ bias,
    const float* __restrict__ Xres, float* __restrict__ Hout)
{
  const int idx = blockIdx.x * 256 + threadIdx.x;
  if (idx >= N_ * F_) return;
  const int col = idx & (F_ - 1);
  Hout[idx] = elu1(T[idx] + bias[col]) + Xres[idx];
}

// ---------------------------------------------------------------------------
extern "C" void kernel_launch(void* const* d_in, const int* in_sizes, int n_in,
                              void* d_out, int out_size, void* d_ws, size_t ws_size,
                              hipStream_t stream) {
  (void)in_sizes; (void)n_in; (void)out_size; (void)ws_size;
  const float* X        = (const float*)d_in[0];
  const float* localp   = (const float*)d_in[1];
  const float* longp    = (const float*)d_in[2];
  const float* W        = (const float*)d_in[3];
  const float* V        = (const float*)d_in[4];
  const float* enc_g    = (const float*)d_in[5];
  const float* enc_be   = (const float*)d_in[6];
  const float* emlp_W   = (const float*)d_in[7];
  const float* emlp_b   = (const float*)d_in[8];
  const float* emlp_g   = (const float*)d_in[9];
  const float* emlp_be  = (const float*)d_in[10];
  const float* dmlp_W   = (const float*)d_in[11];
  const float* dmlp_b   = (const float*)d_in[12];
  const float* dmlp_g   = (const float*)d_in[13];
  const float* dmlp_be  = (const float*)d_in[14];

  char* ws = (char*)d_ws;
  size_t off = 0;
  auto alloc = [&](size_t bytes) -> void* {
    void* p = ws + off;
    off = (off + bytes + 255) & ~(size_t)255;
    return p;
  };
  _Float16* att16 = (_Float16*)alloc((size_t)H_ * N_ * N_ * sizeof(_Float16));
  float*    Xw32  = (float*)   alloc((size_t)N_ * F_ * sizeof(float));
  _Float16* Xw16  = (_Float16*)alloc((size_t)N_ * F_ * sizeof(_Float16));
  float*    f1    = (float*)   alloc((size_t)H_ * N_ * sizeof(float));
  float*    f2    = (float*)   alloc((size_t)H_ * N_ * sizeof(float));
  float*    Xc    = (float*)   alloc((size_t)N_ * F_ * sizeof(float));
  float*    Xn    = (float*)   alloc((size_t)N_ * F_ * sizeof(float));
  _Float16* Xn16  = (_Float16*)alloc((size_t)N_ * F_ * sizeof(_Float16));
  float*    t32   = (float*)   alloc((size_t)N_ * F_ * sizeof(float));
  float*    h32   = (float*)   alloc((size_t)N_ * F_ * sizeof(float));
  float*    emb   = (float*)   alloc((size_t)N_ * F_ * sizeof(float));
  _Float16* emb16 = (_Float16*)alloc((size_t)N_ * F_ * sizeof(_Float16));
  float*    Ya    = (float*)   alloc((size_t)N_ * F_ * sizeof(float));
  float*    Yc    = (float*)   alloc((size_t)N_ * F_ * sizeof(float));
  _Float16* Yc16  = (_Float16*)alloc((size_t)N_ * F_ * sizeof(_Float16));
  float*    mean  = (float*)   alloc((size_t)F_ * sizeof(float));
  float*    rstd  = (float*)   alloc((size_t)F_ * sizeof(float));

  const dim3 blk(256);
  const dim3 gHead(N_ / 128, H_);
  const dim3 gFF(N_ / 128, F_ / 128);
  const int  gEW = (N_ * F_) / 256;

  // ---- Encoder ----
  k_headgemm<<<gHead, blk, 0, stream>>>(X, W, Xw32, Xw16, /*stageTrans=*/1, /*elu=*/0);
  k_fvec<<<(H_ * N_ + 255) / 256, blk, 0, stream>>>(Xw32, V, f1, f2);
  k_scores<<<N_, blk, 0, stream>>>(f1, f2, localp, longp, att16);
  k_attmm<<<gHead, blk, 0, stream>>>(att16, Xw16, Xc);
  k_bn_stats<<<F_, blk, 0, stream>>>(Xc, mean, rstd);
  k_bn_apply<<<gEW, blk, 0, stream>>>(Xc, mean, rstd, enc_g, enc_be, Xn, Xn16);
  k_ffgemm<<<gFF, blk, 0, stream>>>(Xn16, emlp_W, t32);
  k_elu_res<<<gEW, blk, 0, stream>>>(t32, emlp_b, Xn, h32);
  k_bn_stats<<<F_, blk, 0, stream>>>(h32, mean, rstd);
  k_bn_apply<<<gEW, blk, 0, stream>>>(h32, mean, rstd, emlp_g, emlp_be, emb, emb16);

  // ---- Decoder ----
  k_attmm<<<gHead, blk, 0, stream>>>(att16, emb16, Ya);
  k_headgemm<<<gHead, blk, 0, stream>>>(Ya, W, Yc, Yc16, /*stageTrans=*/0, /*elu=*/1);
  k_ffgemm<<<gFF, blk, 0, stream>>>(Yc16, dmlp_W, t32);
  k_elu_res<<<gEW, blk, 0, stream>>>(t32, dmlp_b, Yc, h32);
  k_bn_stats<<<F_, blk, 0, stream>>>(h32, mean, rstd);
  k_bn_apply<<<gEW, blk, 0, stream>>>(h32, mean, rstd, dmlp_g, dmlp_be,
                                      (float*)d_out, (_Float16*)nullptr);
}
